// GCNet_17798344475022
// MI455X (gfx1250) — matrix operand order repaired and checked by reading, not compiled, not run on hardware
//
#include <hip/hip_runtime.h>

typedef unsigned short u16;
typedef __bf16 bf16_t;
typedef bf16_t v16bf __attribute__((ext_vector_type(16)));
typedef float v8f __attribute__((ext_vector_type(8)));
typedef __attribute__((address_space(3))) u16 lds_u16;   // LDS-qualified

union Frag { uint4 u[2]; v16bf v; };

__device__ __forceinline__ u16 f2bf(float f) {
  unsigned u = __float_as_uint(f);
  u += 0x7fffu + ((u >> 16) & 1u);           // round-to-nearest-even
  return (u16)(u >> 16);
}
__device__ __forceinline__ float bf2f(u16 h) {
  return __uint_as_float(((unsigned)h) << 16);
}

// ---------------------------------------------------------------------------
// Implicit-GEMM 3x3 conv (pad=0; padded inputs carry an explicit zero halo),
// channels-last (img,H,W,C) bf16 activations, fp32 acc, WMMA 16x16x32 bf16.
// Block = 256 threads = 8 wave32. Tile: 64(M) x 128(N).
// K spatial-major: k = (ky*3+kx)*Cin + ci -> weight rows are K-contiguous.
//  - A tiles staged DIRECTLY into LDS with global_load_async_to_lds_b32
//    (per-lane gather, no VGPR round trip). The LDS operand is the ptrtoint
//    of an addrspace(3) pointer into As: this escapes/captures As so the
//    "memory" clobber forces real, ordered ds_loads of the staged data
//    (and gives the exact hardware LDS offset, no base-0 assumption).
//  - double-buffered, one s_wait_asynccnt + one barrier per 32-wide K chunk;
//    staging of chunk q+1 overlaps the 4 WMMAs of chunk q
//  - 4 independent A fragments -> ds_loads batch ahead of the WMMA burst
// Requires: M % 64 == 0, Cout % 128 == 0, Cin % 32 == 0  (all true here).
// ---------------------------------------------------------------------------
__global__ __launch_bounds__(256)
void conv3x3_wmma_bf16(const u16* __restrict__ A, const u16* __restrict__ Wt,
                       const float* __restrict__ bias,
                       const float* __restrict__ bng, const float* __restrict__ bnb,
                       u16* __restrict__ Out,
                       int nImg, int Cin, int H, int W, int Cout, int doRelu)
{
  __shared__ __align__(16) u16 As[2][64 * 32];   // double-buffered A tile
  const int Hout = H - 2;
  const int Wout = W - 2;
  const int HWo = Hout * Wout;
  const int Ktot = Cin * 9;
  const int nChunks = Ktot / 32;
  const int mBase = blockIdx.x * 64;
  const int coutBase = blockIdx.y * 128;
  const int tid = threadIdx.x;
  const int lane = tid & 31;
  const int wv = tid >> 5;

  // --- staging assignment: thread covers rows rowT+{0,16,32,48}, k-pair colT ---
  const int rowT = tid >> 4;            // 0..15
  const int colT = (tid & 15) * 2;      // 0..30
  const u16* pbase[4];
#pragma unroll
  for (int r = 0; r < 4; ++r) {
    int m = mBase + rowT + r * 16;
    int img = m / HWo;
    int pix = m - img * HWo;
    int oy = pix / Wout;
    int ox = pix - oy * Wout;
    pbase[r] = A + ((size_t)img * H * W + (size_t)oy * W + ox) * Cin + colT;
  }

  // rolling chunk state (uniform across threads): next chunk = (tap sN, cb cbN)
  int sN = 0, cbN = 0, tapOff = 0;
  auto stage = [&](int bufSel) {
#pragma unroll
    for (int r = 0; r < 4; ++r) {
      const u16* src = pbase[r] + tapOff + cbN;
      // ptrtoint of addrspace(3) pointer == hardware LDS byte offset;
      // escapes As so the asm's memory clobber covers the staged write.
      unsigned loff = (unsigned)(unsigned long long)
          (lds_u16*)&As[bufSel][(rowT + r * 16) * 32 + colT];
      asm volatile("global_load_async_to_lds_b32 %0, %1, off"
                   :: "v"(loff), "v"(src) : "memory");
    }
    cbN += 32;
    if (cbN == Cin) {                  // advance filter tap
      cbN = 0;
      ++sN;
      tapOff += ((sN % 3) == 0) ? (W - 2) * Cin : Cin;
    }
  };

  // --- WMMA operand addressing (ISA 7.12.2 lane layouts) ---
  const int kHalf = lane >> 4;                       // low/high K half of wave
  const int nW = coutBase + wv * 16 + (lane & 15);   // output channel
  const u16* Brow = Wt + (size_t)nW * Ktot + kHalf * 8;
  const u16* ap = &As[0][(lane & 15) * 32 + kHalf * 8];

  v8f acc0 = {}, acc1 = {}, acc2 = {}, acc3 = {};

  stage(0);                                          // prologue: chunk 0
  asm volatile("s_wait_asynccnt 0x0" ::: "memory");
  __syncthreads();

  for (int q = 0; q < nChunks; ++q) {
    const int pb = q & 1;
    const u16* apb = ap + pb * (64 * 32);
    Frag fa0, fa1, fa2, fa3, fb;
    fa0.u[0] = *(const uint4*)(apb);
    fa0.u[1] = *(const uint4*)(apb + 16);
    fa1.u[0] = *(const uint4*)(apb + 512);
    fa1.u[1] = *(const uint4*)(apb + 528);
    fa2.u[0] = *(const uint4*)(apb + 1024);
    fa2.u[1] = *(const uint4*)(apb + 1040);
    fa3.u[0] = *(const uint4*)(apb + 1536);
    fa3.u[1] = *(const uint4*)(apb + 1552);
    const u16* bp = Brow + q * 32;
    fb.u[0] = *(const uint4*)(bp);
    fb.u[1] = *(const uint4*)(bp + 16);
    if (q + 1 < nChunks) stage(pb ^ 1);              // overlap with WMMAs
    acc0 = __builtin_amdgcn_wmma_f32_16x16x32_bf16(false, fa0.v, false, fb.v,
                                                   (short)0, acc0, false, false);
    acc1 = __builtin_amdgcn_wmma_f32_16x16x32_bf16(false, fa1.v, false, fb.v,
                                                   (short)0, acc1, false, false);
    acc2 = __builtin_amdgcn_wmma_f32_16x16x32_bf16(false, fa2.v, false, fb.v,
                                                   (short)0, acc2, false, false);
    acc3 = __builtin_amdgcn_wmma_f32_16x16x32_bf16(false, fa3.v, false, fb.v,
                                                   (short)0, acc3, false, false);
    asm volatile("s_wait_asynccnt 0x0" ::: "memory");
    __syncthreads();
  }

  // --- fused epilogue: bias (+ optional BN affine) (+ optional ReLU) ---
  const float b  = bias ? bias[nW] : 0.f;
  const float g  = bng ? bng[nW] : 1.f;
  const float b2 = bnb ? bnb[nW] : 0.f;
  auto epi = [&](int t, v8f av) {
#pragma unroll
    for (int i = 0; i < 8; ++i) {
      int mRow = mBase + t * 16 + kHalf * 8 + i;  // VGPR i -> M=i / M=8+i
      float v = (av[i] + b) * g + b2;
      if (doRelu) v = fmaxf(v, 0.f);
      Out[(size_t)mRow * Cout + nW] = f2bf(v);    // channels-last, coalesced
    }
  };
  epi(0, acc0);
  epi(1, acc1);
  epi(2, acc2);
  epi(3, acc3);
}

// ---------------------------------------------------------------------------
// Weight repack: w(O,I,3,3) fp32 -> bf16 [O][k], k = (ky*3+kx)*Cin + ci
// ---------------------------------------------------------------------------
__global__ void k_repack_w(const float* __restrict__ src, u16* __restrict__ dst,
                           int Cout, int Cin) {
  const int K = Cin * 9;
  const long total = (long)Cout * K;
  long i = (long)blockIdx.x * blockDim.x + threadIdx.x;
  const long st = (long)gridDim.x * blockDim.x;
  for (; i < total; i += st) {
    int n = (int)(i / K);
    int r = (int)(i - (long)n * K);
    int s = r / Cin;
    int ci = r - s * Cin;
    dst[i] = f2bf(src[((long)n * Cin + ci) * 9 + s]);
  }
}

// x (192,256,11,11) NCHW fp32 -> channels-last bf16 (192,121,256)
__global__ void k_x_to_cl(const float* __restrict__ src, u16* __restrict__ dst) {
  const long total = 192L * 121 * 256;
  long i = (long)blockIdx.x * blockDim.x + threadIdx.x;
  const long st = (long)gridDim.x * blockDim.x;
  for (; i < total; i += st) {
    int img = (int)(i / (121 * 256));
    int r = (int)(i - (long)img * (121 * 256));
    int p = r / 256;
    int c = r - p * 256;
    dst[i] = f2bf(src[((long)img * 256 + c) * 121 + p]);
  }
}

// ---------------------------------------------------------------------------
// Plumbing kernels (channels-last; 7x7=49 interior, 9x9=81 with halo)
// ---------------------------------------------------------------------------
#define SQ 49

__global__ void k_relu_bf16(const u16* __restrict__ s, u16* __restrict__ d, int n) {
  int i = blockIdx.x * blockDim.x + threadIdx.x;
  int st = gridDim.x * blockDim.x;
  for (; i < n; i += st) d[i] = (s[i] & 0x8000u) ? (u16)0 : s[i];
}

// edge_attr(E,7) -> rel(E,32) = relu(xyzlogq @ prd_w^T + prd_b)
__global__ void k_edge_attr(const float* __restrict__ ea, const float* __restrict__ prdw,
                            const float* __restrict__ prdb, float* __restrict__ rel, int E) {
  int e = blockIdx.x * blockDim.x + threadIdx.x;
  if (e >= E) return;
  float qw = ea[e*7+3], vx = ea[e*7+4], vy = ea[e*7+5], vz = ea[e*7+6];
  float w = fminf(fmaxf(qw, -1.f), 1.f);
  float n = sqrtf(vx*vx + vy*vy + vz*vz);
  float s = acosf(w) / fmaxf(n, 1e-8f);
  float a6[6] = { ea[e*7+0], ea[e*7+1], ea[e*7+2], vx*s, vy*s, vz*s };
  for (int j = 0; j < 32; ++j) {
    float acc = prdb[j];
    for (int k = 0; k < 6; ++k) acc += a6[k] * prdw[j*6 + k];
    rel[e*32 + j] = fmaxf(acc, 0.f);
  }
}

// dst(E, P, Ctot): [h[src] | h[dst] | (ef) | rel], P=81 (zero halo) or 49.
// One thread per (e,pixel); vectorized uint4 row copies.
__global__ void k_concat_edge(u16* __restrict__ dst, const u16* __restrict__ h,
                              const u16* __restrict__ ef, const float* __restrict__ rel,
                              const int* __restrict__ srcI, const int* __restrict__ dstI,
                              int E, int useEf, int halo) {
  const int Ctot = useEf ? 416 : 288;
  const int P = halo ? 81 : SQ;
  int t = blockIdx.x * blockDim.x + threadIdx.x;
  if (t >= E * P) return;
  int e = t / P;
  int p = t - e * P;
  u16* drow = dst + (size_t)t * Ctot;
  uint4* d4 = (uint4*)drow;
  int s = p;
  if (halo) {
    int y = p / 9, x2 = p - y * 9;
    if (y < 1 || y > 7 || x2 < 1 || x2 > 7) {      // zero border
      uint4 z = {0, 0, 0, 0};
      int nseg = Ctot / 8;
      for (int j = 0; j < nseg; ++j) d4[j] = z;
      return;
    }
    s = (y - 1) * 7 + (x2 - 1);
  }
  const uint4* hs = (const uint4*)(h + ((size_t)srcI[e] * SQ + s) * 128);
  const uint4* hd = (const uint4*)(h + ((size_t)dstI[e] * SQ + s) * 128);
#pragma unroll
  for (int j = 0; j < 16; ++j) d4[j] = hs[j];
#pragma unroll
  for (int j = 0; j < 16; ++j) d4[16 + j] = hd[j];
  int coff = 256;
  if (useEf) {
    const uint4* ep = (const uint4*)(ef + ((size_t)e * SQ + s) * 128);
#pragma unroll
    for (int j = 0; j < 16; ++j) d4[32 + j] = ep[j];
    coff = 384;
  }
  const float* rp = rel + e * 32;
#pragma unroll
  for (int j = 0; j < 32; ++j) drow[coff + j] = f2bf(rp[j]);
}

// dst(E,81,256): [h[src] | e_new] with zero halo
__global__ void k_concat_msg(u16* __restrict__ dst, const u16* __restrict__ h,
                             const int* __restrict__ srcI, const u16* __restrict__ enew,
                             int E) {
  int t = blockIdx.x * blockDim.x + threadIdx.x;
  if (t >= E * 81) return;
  int e = t / 81;
  int p = t - e * 81;
  uint4* d4 = (uint4*)(dst + (size_t)t * 256);
  int y = p / 9, x2 = p - y * 9;
  if (y < 1 || y > 7 || x2 < 1 || x2 > 7) {
    uint4 z = {0, 0, 0, 0};
#pragma unroll
    for (int j = 0; j < 32; ++j) d4[j] = z;
    return;
  }
  int s = (y - 1) * 7 + (x2 - 1);
  const uint4* hs = (const uint4*)(h + ((size_t)srcI[e] * SQ + s) * 128);
  const uint4* en = (const uint4*)(enew + ((size_t)e * SQ + s) * 128);
#pragma unroll
  for (int j = 0; j < 16; ++j) d4[j] = hs[j];
#pragma unroll
  for (int j = 0; j < 16; ++j) d4[16 + j] = en[j];
}

// dst(N,81,256): [h | aggr] with zero halo
__global__ void k_concat_upd(u16* __restrict__ dst, const u16* __restrict__ h,
                             const u16* __restrict__ aggr, int N) {
  int t = blockIdx.x * blockDim.x + threadIdx.x;
  if (t >= N * 81) return;
  int n = t / 81;
  int p = t - n * 81;
  uint4* d4 = (uint4*)(dst + (size_t)t * 256);
  int y = p / 9, x2 = p - y * 9;
  if (y < 1 || y > 7 || x2 < 1 || x2 > 7) {
    uint4 z = {0, 0, 0, 0};
#pragma unroll
    for (int j = 0; j < 32; ++j) d4[j] = z;
    return;
  }
  int s = (y - 1) * 7 + (x2 - 1);
  const uint4* a = (const uint4*)(h + ((size_t)n * SQ + s) * 128);
  const uint4* b = (const uint4*)(aggr + ((size_t)n * SQ + s) * 128);
#pragma unroll
  for (int j = 0; j < 16; ++j) d4[j] = a[j];
#pragma unroll
  for (int j = 0; j < 16; ++j) d4[16 + j] = b[j];
}

// aggr[node] = mean over 5 incoming edges (complete-minus-self graph, npg=6)
__global__ void k_aggregate(const u16* __restrict__ msg, u16* __restrict__ aggr) {
  const long cs = 128L * SQ;
  const long total = 192L * cs;
  long i = (long)blockIdx.x * blockDim.x + threadIdx.x;
  const long st = (long)gridDim.x * blockDim.x;
  for (; i < total; i += st) {
    int n = (int)(i / cs);
    long r = i - (long)n * cs;
    int g = n / 6, di = n - g * 6;
    float acc = 0.f;
    for (int si = 0; si < 6; ++si) {
      if (si == di) continue;
      int el = si * 5 + (di < si ? di : di - 1);
      acc += bf2f(msg[((long)(g * 30 + el)) * cs + r]);
    }
    aggr[i] = f2bf(acc * 0.2f);
  }
}

// rel_pose_out(E,6): er2 is channels-last (e, 9, 128); weights are (3,128,3,3)
__global__ void k_head6(const u16* __restrict__ er2, const float* __restrict__ xrw,
                        const float* __restrict__ xrb, const float* __restrict__ lrw,
                        const float* __restrict__ lrb, float* __restrict__ out, int E) {
  int t = blockIdx.x * blockDim.x + threadIdx.x;
  if (t >= E * 6) return;
  int e = t / 6, j = t - e * 6;
  const float* wrow = (j < 3) ? (xrw + j * 1152) : (lrw + (j - 3) * 1152);
  float acc = (j < 3) ? xrb[j] : lrb[j - 3];
  const u16* a = er2 + (long)e * 1152;
  for (int s = 0; s < 9; ++s)
    for (int c = 0; c < 128; ++c)
      acc += bf2f(a[s * 128 + c]) * wrow[c * 9 + s];
  out[e * 6 + j] = acc;
}

// per-graph attention: softmax(q k^T / sqrt(D)) v, D=1152 (dot order-invariant)
__global__ __launch_bounds__(256)
void k_attn1(const u16* __restrict__ q, const u16* __restrict__ k,
             const u16* __restrict__ v, float* __restrict__ he) {
  __shared__ float sc[30 * 30];
  const int g = blockIdx.x, tid = threadIdx.x;
  const float scale = rsqrtf(1152.f);
  for (int p = tid; p < 900; p += 256) {
    int e = p / 30, f = p - (p / 30) * 30;
    const u16* qp = q + (long)(g * 30 + e) * 1152;
    const u16* kp = k + (long)(g * 30 + f) * 1152;
    float acc = 0.f;
    for (int d = 0; d < 1152; ++d) acc += bf2f(qp[d]) * bf2f(kp[d]);
    sc[p] = acc * scale;
  }
  __syncthreads();
  if (tid < 30) {
    float mx = -1e30f;
    for (int f = 0; f < 30; ++f) mx = fmaxf(mx, sc[tid*30 + f]);
    float sum = 0.f;
    for (int f = 0; f < 30; ++f) { float w = __expf(sc[tid*30+f] - mx); sc[tid*30+f] = w; sum += w; }
    float inv = 1.f / sum;
    for (int f = 0; f < 30; ++f) sc[tid*30 + f] *= inv;
  }
  __syncthreads();
  for (int p = tid; p < 30 * 1152; p += 256) {
    int e = p / 1152, d = p - e * 1152;
    float acc = 0.f;
    for (int f = 0; f < 30; ++f) acc += sc[e*30 + f] * bf2f(v[(long)(g*30 + f) * 1152 + d]);
    he[(long)(g * 30 + e) * 1152 + d] = acc;   // (s,c) channel-last order
  }
}

// per-graph: edge-logit softmax pooling -> xh/lh heads -> qexp -> SE(3) matrix
__global__ __launch_bounds__(256)
void k_attn2(const float* __restrict__ he, const float* __restrict__ eaw,
             const float* __restrict__ eab, const float* __restrict__ xhw,
             const float* __restrict__ xhb, const float* __restrict__ lhw,
             const float* __restrict__ lhb, float* __restrict__ outM) {
  __shared__ float wgt[30];
  __shared__ float agg[1152];
  __shared__ float head[6];
  const int g = blockIdx.x, tid = threadIdx.x;
  if (tid < 30) {
    const float* hp = he + (long)(g * 30 + tid) * 1152;
    float acc = eab[0];
    for (int s = 0; s < 9; ++s)
      for (int c = 0; c < 128; ++c) acc += hp[s * 128 + c] * eaw[c * 9 + s];
    wgt[tid] = acc;
  }
  __syncthreads();
  if (tid == 0) {
    float mx = -1e30f;
    for (int e = 0; e < 30; ++e) mx = fmaxf(mx, wgt[e]);
    float sum = 0.f;
    for (int e = 0; e < 30; ++e) { wgt[e] = __expf(wgt[e] - mx); sum += wgt[e]; }
    float inv = 1.f / sum;
    for (int e = 0; e < 30; ++e) wgt[e] *= inv;
  }
  __syncthreads();
  for (int d = tid; d < 1152; d += 256) {
    float acc = 0.f;
    for (int e = 0; e < 30; ++e) acc += wgt[e] * he[(long)(g*30 + e) * 1152 + d];
    agg[d] = acc;                               // (s,c) order
  }
  __syncthreads();
  if (tid < 6) {
    const float* wrow = (tid < 3) ? (xhw + tid * 1152) : (lhw + (tid - 3) * 1152);
    float acc = (tid < 3) ? xhb[tid] : lhb[tid - 3];
    for (int s = 0; s < 9; ++s)
      for (int c = 0; c < 128; ++c) acc += agg[s * 128 + c] * wrow[c * 9 + s];
    head[tid] = acc;
  }
  __syncthreads();
  if (tid == 0) {
    float tx = head[0], ty = head[1], tz = head[2];
    float vx = head[3], vy = head[4], vz = head[5];
    float n = sqrtf(vx*vx + vy*vy + vz*vz);
    float w = cosf(n);
    float sc = sinf(n) / fmaxf(n, 1e-8f);
    float x = vx * sc, y = vy * sc, z = vz * sc;
    float* o = outM + g * 16;
    o[0]  = 1.f - 2.f*(y*y + z*z); o[1]  = 2.f*(x*y - w*z); o[2]  = 2.f*(x*z + w*y); o[3]  = tx;
    o[4]  = 2.f*(x*y + w*z); o[5]  = 1.f - 2.f*(x*x + z*z); o[6]  = 2.f*(y*z - w*x); o[7]  = ty;
    o[8]  = 2.f*(x*z - w*y); o[9]  = 2.f*(y*z + w*x); o[10] = 1.f - 2.f*(x*x + y*y); o[11] = tz;
    o[12] = 0.f; o[13] = 0.f; o[14] = 0.f; o[15] = 1.f;
  }
}

__global__ void k_ei_out(const int* __restrict__ ei, float* __restrict__ out, int n) {
  int i = blockIdx.x * blockDim.x + threadIdx.x;
  if (i < n) out[i] = (float)ei[i];
}

// ---------------------------------------------------------------------------
// Host orchestration
// ---------------------------------------------------------------------------
static inline int ceil_div(long a, int b) { return (int)((a + b - 1) / b); }

extern "C" void kernel_launch(void* const* d_in, const int* in_sizes, int n_in,
                              void* d_out, int out_size, void* d_ws, size_t ws_size,
                              hipStream_t stream) {
  (void)in_sizes; (void)n_in; (void)out_size; (void)ws_size;
  // --- inputs (setup_inputs order; params in dict-insertion order) ---
  const float* x          = (const float*)d_in[0];   // (192,256,11,11)
  const float* edge_attr  = (const float*)d_in[1];   // (960,7)
  const int*   edge_index = (const int*)  d_in[2];   // (2,960)
  const float* pf_w1   = (const float*)d_in[5];  const float* pf_w1_b = (const float*)d_in[6];
  const float* pf_w2   = (const float*)d_in[7];  const float* pf_w2_b = (const float*)d_in[8];
  const float* pie_w   = (const float*)d_in[9];  const float* pie_b   = (const float*)d_in[10];
  const float* we_w    = (const float*)d_in[11]; const float* we_b    = (const float*)d_in[12];
  const float* wm_w    = (const float*)d_in[13]; const float* wm_b    = (const float*)d_in[14];
  const float* wu_w    = (const float*)d_in[15]; const float* wu_b    = (const float*)d_in[16];
  const float* er1_w   = (const float*)d_in[17]; const float* er1_b   = (const float*)d_in[18];
  const float* er2_w   = (const float*)d_in[19]; const float* er2_b   = (const float*)d_in[20];
  const float* xr_w    = (const float*)d_in[21]; const float* xr_b    = (const float*)d_in[22];
  const float* lr_w    = (const float*)d_in[23]; const float* lr_b    = (const float*)d_in[24];
  const float* q1_w    = (const float*)d_in[25]; const float* q1_b    = (const float*)d_in[26];
  const float* q2_w    = (const float*)d_in[27]; const float* q2_b    = (const float*)d_in[28];
  const float* k1_w    = (const float*)d_in[29]; const float* k1_b    = (const float*)d_in[30];
  const float* k2_w    = (const float*)d_in[31]; const float* k2_b    = (const float*)d_in[32];
  const float* v1_w    = (const float*)d_in[33]; const float* v1_b    = (const float*)d_in[34];
  const float* v2_w    = (const float*)d_in[35]; const float* v2_b    = (const float*)d_in[36];
  const float* ea_w    = (const float*)d_in[37]; const float* ea_b    = (const float*)d_in[38];
  const float* xh_w    = (const float*)d_in[39]; const float* xh_b    = (const float*)d_in[40];
  const float* lh_w    = (const float*)d_in[41]; const float* lh_b    = (const float*)d_in[42];
  const float* prd_w   = (const float*)d_in[43]; const float* prd_b   = (const float*)d_in[44];
  const float* bn_g    = (const float*)d_in[45]; const float* bn_b    = (const float*)d_in[46];

  const int E = 960;
  const int* srcI = edge_index;
  const int* dstI = edge_index + E;

  // --- workspace bump allocator ---
  char* base = (char*)d_ws;
  size_t off = 0;
  auto alloc = [&](size_t bytes) -> void* {
    off = (off + 255) & ~(size_t)255;
    void* p = base + off;
    off += bytes;
    return p;
  };
  auto allocH = [&](size_t n) -> u16*   { return (u16*)  alloc(n * 2); };
  auto allocF = [&](size_t n) -> float* { return (float*)alloc(n * 4); };

  // bf16 weight repacks (spatial-major K)
  struct { const float* s; u16* d; int co; int ci; } wc[14] = {
    { pf_w1, nullptr, 256, 256 }, { pf_w2, nullptr, 128, 256 },
    { pie_w, nullptr, 128, 288 }, { we_w,  nullptr, 128, 416 },
    { wm_w,  nullptr, 128, 256 }, { wu_w,  nullptr, 128, 256 },
    { er1_w, nullptr, 128, 128 }, { er2_w, nullptr, 128, 128 },
    { q1_w,  nullptr, 128, 416 }, { q2_w,  nullptr, 128, 128 },
    { k1_w,  nullptr, 128, 416 }, { k2_w,  nullptr, 128, 128 },
    { v1_w,  nullptr, 128, 416 }, { v2_w,  nullptr, 128, 128 } };
  for (int i = 0; i < 14; ++i) wc[i].d = allocH((size_t)wc[i].co * wc[i].ci * 9);
  u16 *wb_pf1 = wc[0].d, *wb_pf2 = wc[1].d, *wb_pie = wc[2].d, *wb_we = wc[3].d,
      *wb_wm = wc[4].d, *wb_wu = wc[5].d, *wb_er1 = wc[6].d, *wb_er2 = wc[7].d,
      *wb_q1 = wc[8].d, *wb_q2 = wc[9].d, *wb_k1 = wc[10].d, *wb_k2 = wc[11].d,
      *wb_v1 = wc[12].d, *wb_v2 = wc[13].d;

  // channels-last activation buffers (halo buffers are 9x9 spatial)
  u16* x_bf  = allocH(192L*121*256);
  u16* h1    = allocH(192L*81*256);
  u16* h     = allocH(192L*49*128);
  u16* cat   = allocH((size_t)E*81*416);     // shared concat scratch (largest)
  u16* ef    = allocH((size_t)E*49*128);
  u16* enew  = allocH((size_t)E*49*128);
  u16* msg   = allocH((size_t)E*49*128);
  u16* aggr  = allocH(192L*49*128);
  u16* tmp5  = allocH((size_t)E*25*128);     // 5x5 intermediates (er1, q1, k1, v1)
  u16* er2b  = allocH((size_t)E*9*128);
  u16* q2b   = allocH((size_t)E*9*128);
  u16* k2b   = allocH((size_t)E*9*128);
  u16* v2b   = allocH((size_t)E*9*128);
  float* rel = allocF((size_t)E*32);
  float* heF = allocF(32L*30*1152);

  float* outM    = (float*)d_out;            // (32,4,4)
  float* outRel  = outM + 512;               // (960,6)
  float* outEdge = outRel + 5760;            // (2,960)

  auto ew = [&](long n) { return dim3(ceil_div(n, 256)); };

  // --- weight repack + input conversion ---
  for (int i = 0; i < 14; ++i)
    hipLaunchKernelGGL(k_repack_w, ew((long)wc[i].co * wc[i].ci * 9), dim3(256), 0, stream,
                       wc[i].s, wc[i].d, wc[i].co, wc[i].ci);
  hipLaunchKernelGGL(k_x_to_cl, ew(192L*121*256), dim3(256), 0, stream, x, x_bf);
  hipLaunchKernelGGL(k_edge_attr, ew(E), dim3(256), 0, stream, edge_attr, prd_w, prd_b, rel, E);

  auto conv = [&](const u16* Ain, const u16* Wb, const float* bias,
                  const float* g, const float* bb, u16* Outp,
                  int nImg, int Cin, int H, int W, int Cout, int relu) {
    dim3 grid(nImg * (H - 2) * (W - 2) / 64, Cout / 128);
    hipLaunchKernelGGL(conv3x3_wmma_bf16, grid, dim3(256), 0, stream,
                       Ain, Wb, bias, g, bb, Outp, nImg, Cin, H, W, Cout, relu);
  };

  // --- point feature extractor: relu(bn(conv)) then conv ---
  conv(x_bf, wb_pf1, pf_w1_b, bn_g, bn_b, h1, 192, 256, 11, 11, 256, 1);
  conv(h1,   wb_pf2, pf_w2_b, nullptr, nullptr, h, 192, 256, 9, 9, 128, 0);

  // --- pairwise interaction encoder (haloed input -> pad=1 semantics) ---
  hipLaunchKernelGGL(k_concat_edge, ew((long)E*81), dim3(256), 0, stream,
                     cat, h, (const u16*)nullptr, rel, srcI, dstI, E, 0, 1);
  conv(cat, wb_pie, pie_b, nullptr, nullptr, ef, E, 288, 9, 9, 128, 1);

  // --- 2 GNN rounds ---
  for (int it = 0; it < 2; ++it) {
    hipLaunchKernelGGL(k_concat_edge, ew((long)E*81), dim3(256), 0, stream,
                       cat, h, ef, rel, srcI, dstI, E, 1, 1);
    conv(cat, wb_we, we_b, nullptr, nullptr, enew, E, 416, 9, 9, 128, 0);
    hipLaunchKernelGGL(k_concat_msg, ew((long)E*81), dim3(256), 0, stream,
                       cat, h, srcI, enew, E);
    conv(cat, wb_wm, wm_b, nullptr, nullptr, msg, E, 256, 9, 9, 128, 0);
    hipLaunchKernelGGL(k_aggregate, ew(192L*128*SQ), dim3(256), 0, stream, msg, aggr);
    hipLaunchKernelGGL(k_concat_upd, ew(192L*81), dim3(256), 0, stream,
                       cat, h, aggr, 192);
    conv(cat, wb_wu, wu_b, nullptr, nullptr, h, 192, 256, 9, 9, 128, 1);
    hipLaunchKernelGGL(k_relu_bf16, ew((long)E*128*SQ), dim3(256), 0, stream,
                       enew, ef, E*128*SQ);
  }

  // --- edge regression head: 7x7 -> 5x5 -> 3x3 -> (xyz,logq) ---
  conv(ef,   wb_er1, er1_b, nullptr, nullptr, tmp5, E, 128, 7, 7, 128, 1);
  conv(tmp5, wb_er2, er2_b, nullptr, nullptr, er2b, E, 128, 5, 5, 128, 1);
  hipLaunchKernelGGL(k_head6, ew(E*6), dim3(256), 0, stream,
                     er2b, xr_w, xr_b, lr_w, lr_b, outRel, E);

  // --- attention q/k/v nets (7x7 input, true pad=0) ---
  hipLaunchKernelGGL(k_concat_edge, ew((long)E*SQ), dim3(256), 0, stream,
                     cat, h, ef, rel, srcI, dstI, E, 1, 0);
  conv(cat,  wb_q1, q1_b, nullptr, nullptr, tmp5, E, 416, 7, 7, 128, 1);
  conv(tmp5, wb_q2, q2_b, nullptr, nullptr, q2b,  E, 128, 5, 5, 128, 0);
  conv(cat,  wb_k1, k1_b, nullptr, nullptr, tmp5, E, 416, 7, 7, 128, 1);
  conv(tmp5, wb_k2, k2_b, nullptr, nullptr, k2b,  E, 128, 5, 5, 128, 0);
  conv(cat,  wb_v1, v1_b, nullptr, nullptr, tmp5, E, 416, 7, 7, 128, 1);
  conv(tmp5, wb_v2, v2_b, nullptr, nullptr, v2b,  E, 128, 5, 5, 128, 0);

  // --- per-graph attention + pooling + SE(3) output ---
  hipLaunchKernelGGL(k_attn1, dim3(32), dim3(256), 0, stream, q2b, k2b, v2b, heF);
  hipLaunchKernelGGL(k_attn2, dim3(32), dim3(256), 0, stream,
                     heF, ea_w, ea_b, xh_w, xh_b, lh_w, lh_b, outM);

  // --- edge_index passthrough ---
  hipLaunchKernelGGL(k_ei_out, ew(2*E), dim3(256), 0, stream, edge_index, outEdge, 2*E);
}
